// QSAE_49709951484365
// MI455X (gfx1250) — matrix operand order repaired and checked by reading, not compiled, not run on hardware
//
#include <hip/hip_runtime.h>
#include <hip/hip_bf16.h>

// ---------------------------------------------------------------------------
// QSAE forward for MI455X (gfx1250):
//   bf16 WMMA GEMMs fed by double-buffered GLOBAL_LOAD_ASYNC_TO_LDS_B128
//   + exact per-row radix top-k threshold.
// ---------------------------------------------------------------------------

constexpr int DM = 4096;     // d_model
constexpr int DH = 32768;    // d_hidden
constexpr int NT = 4096;     // tokens
constexpr int KRANK = 513;   // 513-th largest per row (quantile 'lower')
constexpr int ROWP = 36;     // LDS row pitch in floats (bank-conflict-free)

typedef __attribute__((ext_vector_type(16))) __bf16        v16bf;
typedef __attribute__((ext_vector_type(8)))  float         v8f;
typedef __attribute__((ext_vector_type(8)))  unsigned int  v8u;

// ---------------------------------------------------------------------------
// gfx1250 async copy: global -> LDS, 16 bytes per lane, tracked by ASYNCcnt.
// LDS byte offset = low 32 bits of the flat shared address (LDS aperture).
// ---------------------------------------------------------------------------
__device__ __forceinline__ void async_copy_b128(unsigned int lds_byte_off,
                                                const float* gptr) {
    asm volatile("global_load_async_to_lds_b128 %0, %1, off"
                 :: "v"(lds_byte_off), "v"((unsigned long long)(size_t)gptr)
                 : "memory");
}

template <int N>
__device__ __forceinline__ void wait_async() {
#if __has_builtin(__builtin_amdgcn_s_wait_asynccnt)
    __builtin_amdgcn_s_wait_asynccnt(N);
#else
    asm volatile("s_wait_asynccnt %0" :: "n"(N) : "memory");
#endif
}

// fp32 pair -> packed bf16 (one VALU op per pair).
__device__ __forceinline__ unsigned int pack_bf16_pair(float lo, float hi) {
#if __has_builtin(__builtin_amdgcn_cvt_pk_bf16_f32)
    typedef __attribute__((ext_vector_type(2))) __bf16 v2bf;
    v2bf p = __builtin_amdgcn_cvt_pk_bf16_f32(lo, hi);
    return __builtin_bit_cast(unsigned int, p);
#else
    // v_perm_b32 truncation: dst = {hi[31:16], lo[31:16]}
    return __builtin_amdgcn_perm(__float_as_uint(hi), __float_as_uint(lo),
                                 0x07060302u);
#endif
}

__device__ __forceinline__ v16bf pack16(float4 f0, float4 f1, float4 f2, float4 f3) {
    v8u w;
    w[0] = pack_bf16_pair(f0.x, f0.y);
    w[1] = pack_bf16_pair(f0.z, f0.w);
    w[2] = pack_bf16_pair(f1.x, f1.y);
    w[3] = pack_bf16_pair(f1.z, f1.w);
    w[4] = pack_bf16_pair(f2.x, f2.y);
    w[5] = pack_bf16_pair(f2.z, f2.w);
    w[6] = pack_bf16_pair(f3.x, f3.y);
    w[7] = pack_bf16_pair(f3.z, f3.w);
    return __builtin_bit_cast(v16bf, w);
}

// Unused: 16 contiguous-K fp32 at p (LDS) -> ds_load_b128 x4 + pack.
__device__ __forceinline__ v16bf load_frag16(const float* p) {
    float4 f0 = *(const float4*)(p);
    float4 f1 = *(const float4*)(p + 4);
    float4 f2 = *(const float4*)(p + 8);
    float4 f3 = *(const float4*)(p + 12);
    return pack16(f0, f1, f2, f3);
}

// A fragment: two 8-float chunks at p and p+16 (K split by lane half).
__device__ __forceinline__ v16bf load_fragA(const float* p) {
    float4 f0 = *(const float4*)(p);
    float4 f1 = *(const float4*)(p + 4);
    float4 f2 = *(const float4*)(p + 16);
    float4 f3 = *(const float4*)(p + 20);
    return pack16(f0, f1, f2, f3);
}

__device__ __forceinline__ float block_reduce_sum_256(float v) {
    __shared__ float red[256];
    red[threadIdx.x] = v;
    __syncthreads();
    #pragma unroll
    for (int s = 128; s > 0; s >>= 1) {
        if ((int)threadIdx.x < s) red[threadIdx.x] += red[threadIdx.x + s];
        __syncthreads();
    }
    return red[0];
}

// ---------------------------------------------------------------------------
// Async-pipelined bf16 WMMA GEMM core, shared by encode/decode.
//   C[64 x 256] block tile, K-step 32, A rows from `amat` (ldK), B rows from
//   `bmat` (ldK).  8 waves as 2 (n) x 4 (col); wave tile 32x64 = 8 WMMAs/step.
//   Per K-step each thread issues 10 async b128 copies (8 B-tile + 2 A-tile),
//   double buffered: wait asynccnt<=10 keeps the prefetch in flight.
// ---------------------------------------------------------------------------
template <int KDIM>
__device__ __forceinline__ void gemm_tile_async(const float* __restrict__ amat,
                                                const float* __restrict__ bmat,
                                                int n0b, int c0b,
                                                float sB[2][256 * ROWP],
                                                float sA[2][64 * ROWP],
                                                v8f acc[2][4]) {
    const int tid  = threadIdx.x;
    const int lane = tid & 31;
    const int wave = tid >> 5;
    const int lr   = lane & 15;
    const int hi   = lane >> 4;
    const int wn   = (wave & 1) * 32;    // wave token offset in tile
    const int wc   = (wave >> 1) * 64;   // wave column offset in tile

    // Staging assignments: B row = tid (8 chunks); A row = tid&63, 2 chunks.
    const float* gB = bmat + (size_t)(c0b + tid) * KDIM;
    const int arow = tid & 63;
    const int aq   = (tid >> 6) * 2;     // first of two 16B chunks
    const float* gA = amat + (size_t)(n0b + arow) * KDIM + aq * 4;

    unsigned int ldsB[2], ldsA[2];
    ldsB[0] = (unsigned int)(size_t)&sB[0][tid * ROWP];
    ldsB[1] = (unsigned int)(size_t)&sB[1][tid * ROWP];
    ldsA[0] = (unsigned int)(size_t)&sA[0][arow * ROWP + aq * 4];
    ldsA[1] = (unsigned int)(size_t)&sA[1][arow * ROWP + aq * 4];

    auto stage = [&](int buf, int k0) {
        const float* gb = gB + k0;
        unsigned int lb = ldsB[buf];
        #pragma unroll
        for (int q = 0; q < 8; ++q)
            async_copy_b128(lb + (unsigned)(q * 16), gb + q * 4);
        const float* ga = gA + k0;
        async_copy_b128(ldsA[buf], ga);
        async_copy_b128(ldsA[buf] + 16u, ga + 4);
    };

    stage(0, 0);
    int cur = 0;

    for (int k0 = 0; k0 < KDIM; k0 += 32) {
        const bool has_next = (k0 + 32) < KDIM;
        if (has_next) { stage(cur ^ 1, k0 + 32); wait_async<10>(); }
        else          { wait_async<0>(); }
        __syncthreads();

        const float* aBuf = &sA[cur][0];
        const float* bBuf = &sB[cur][0];
        v16bf a0 = load_fragA(aBuf + (wn + lr) * ROWP + hi * 8);
        v16bf a1 = load_fragA(aBuf + (wn + 16 + lr) * ROWP + hi * 8);

        #pragma unroll
        for (int t = 0; t < 4; ++t) {
            v16bf b = load_fragA(bBuf + (wc + t * 16 + lr) * ROWP + hi * 8);
            acc[0][t] = __builtin_amdgcn_wmma_f32_16x16x32_bf16(
                false, a0, false, b, (short)0, acc[0][t], false, false);
            acc[1][t] = __builtin_amdgcn_wmma_f32_16x16x32_bf16(
                false, a1, false, b, (short)0, acc[1][t], false, false);
        }
        __syncthreads();   // all waves done reading before buffer is refilled
        cur ^= 1;
    }
}

// ---------------------------------------------------------------------------
// Encode: scores = sigmoid(x @ W_enc^T + b_enc)
// ---------------------------------------------------------------------------
__global__ __launch_bounds__(256) void qsae_encode(const float* __restrict__ x,
                                                   const float* __restrict__ W_enc,
                                                   const float* __restrict__ b_enc,
                                                   float* __restrict__ scores) {
    __shared__ float sB[2][256 * ROWP];
    __shared__ float sA[2][64 * ROWP];

    const int lane = threadIdx.x & 31;
    const int wave = threadIdx.x >> 5;
    const int lr   = lane & 15;
    const int hi   = lane >> 4;
    const int n0b  = blockIdx.x * 64;
    const int h0b  = blockIdx.y * 256;
    const int n0   = n0b + (wave & 1) * 32;
    const int h0   = h0b + (wave >> 1) * 64;

    v8f acc[2][4] = {};
    gemm_tile_async<DM>(x, W_enc, n0b, h0b, sB, sA, acc);

    // Epilogue: bias + sigmoid.  C/D layout: VGPR r -> M = r + 8*hi, N = lr.
    #pragma unroll
    for (int s = 0; s < 2; ++s) {
        #pragma unroll
        for (int t = 0; t < 4; ++t) {
            const int h = h0 + t * 16 + lr;
            const float bias = b_enc[h];
            #pragma unroll
            for (int r = 0; r < 8; ++r) {
                const int n = n0 + s * 16 + hi * 8 + r;
                float v = acc[s][t][r] + bias;
                scores[(size_t)n * DH + h] = 1.0f / (1.0f + __expf(-v));
            }
        }
    }
}

// ---------------------------------------------------------------------------
// Decode: x_rec = act @ W_dec^T, fused reconstruction-MSE epilogue.
// Binary activations are exact in bf16.
// ---------------------------------------------------------------------------
__global__ __launch_bounds__(256) void qsae_decode(const float* __restrict__ act,
                                                   const float* __restrict__ W_dec,
                                                   const float* __restrict__ x,
                                                   float* __restrict__ xrec,
                                                   float* __restrict__ rerr) {
    __shared__ float sB[2][256 * ROWP];
    __shared__ float sA[2][64 * ROWP];

    const int tid  = threadIdx.x;
    const int lane = tid & 31;
    const int wave = tid >> 5;
    const int lr   = lane & 15;
    const int hi   = lane >> 4;
    const int n0b  = blockIdx.x * 64;
    const int d0b  = blockIdx.y * 256;
    const int n0   = n0b + (wave & 1) * 32;
    const int d0   = d0b + (wave >> 1) * 64;

    v8f acc[2][4] = {};
    gemm_tile_async<DH>(act, W_dec, n0b, d0b, sB, sA, acc);

    float lsum = 0.0f;
    #pragma unroll
    for (int s = 0; s < 2; ++s) {
        #pragma unroll
        for (int t = 0; t < 4; ++t) {
            const int d = d0 + t * 16 + lr;
            #pragma unroll
            for (int r = 0; r < 8; ++r) {
                const int n = n0 + s * 16 + hi * 8 + r;
                float v = acc[s][t][r];
                xrec[(size_t)n * DM + d] = v;
                float diff = v - x[(size_t)n * DM + d];
                lsum += diff * diff;
            }
        }
    }

    // Block reduction reusing the (now idle) staging buffer.
    float* red = &sB[0][0];
    red[tid] = lsum;
    __syncthreads();
    #pragma unroll
    for (int s = 128; s > 0; s >>= 1) {
        if (tid < s) red[tid] += red[tid + s];
        __syncthreads();
    }
    if (tid == 0)
        atomicAdd(rerr, red[0] * (1.0f / ((float)NT * (float)DM)));
}

// ---------------------------------------------------------------------------
// Exact 513-th largest per row: 4-round radix select on float bit patterns
// (sigmoid outputs are positive -> unsigned bit order == float order).
// ---------------------------------------------------------------------------
__global__ __launch_bounds__(256) void qsae_thresh(const float* __restrict__ scores,
                                                   float* __restrict__ thr) {
    __shared__ unsigned int hist[256];
    __shared__ unsigned int sh_prefix;
    __shared__ int sh_krem;

    const int row = blockIdx.x;
    const float* srow = scores + (size_t)row * DH;

    if (threadIdx.x == 0) { sh_prefix = 0u; sh_krem = KRANK; }
    unsigned int prefmask = 0u;

    for (int shift = 24; shift >= 0; shift -= 8) {
        hist[threadIdx.x] = 0u;
        __syncthreads();
        const unsigned int prefix = sh_prefix;

        for (int i = threadIdx.x; i < DH; i += 256) {
            unsigned int u = __float_as_uint(srow[i]);
            if ((u & prefmask) == prefix)
                atomicAdd(&hist[(u >> shift) & 255u], 1u);
        }
        __syncthreads();

        if (threadIdx.x == 0) {
            int k = sh_krem;
            unsigned int cum = 0u;
            int d = 0;
            for (d = 255; d >= 0; --d) {
                unsigned int c = hist[d];
                if (cum + c >= (unsigned int)k) { sh_krem = k - (int)cum; break; }
                cum += c;
            }
            sh_prefix = prefix | ((unsigned int)d << shift);
        }
        __syncthreads();
        prefmask |= (0xFFu << shift);
    }

    if (threadIdx.x == 0) thr[row] = __uint_as_float(sh_prefix);
}

// ---------------------------------------------------------------------------
// activations = (scores >= thr[n]); quantization_error += mean((s - a)^2)
// ---------------------------------------------------------------------------
__global__ __launch_bounds__(256) void qsae_act(const float* __restrict__ scores,
                                                const float* __restrict__ thr,
                                                float* __restrict__ act,
                                                float* __restrict__ qerr) {
    const int row = blockIdx.x;
    const float t = thr[row];
    const size_t base = (size_t)row * DH + (size_t)blockIdx.y * 4096;

    float lsum = 0.0f;
    #pragma unroll
    for (int v = 0; v < 4; ++v) {
        size_t off = base + ((size_t)(v * 256 + threadIdx.x)) * 4;
        float4 s = *(const float4*)(scores + off);
        float4 a;
        a.x = (s.x >= t) ? 1.0f : 0.0f;
        a.y = (s.y >= t) ? 1.0f : 0.0f;
        a.z = (s.z >= t) ? 1.0f : 0.0f;
        a.w = (s.w >= t) ? 1.0f : 0.0f;
        *(float4*)(act + off) = a;
        float dx = s.x - a.x, dy = s.y - a.y, dz = s.z - a.z, dw = s.w - a.w;
        lsum += dx * dx + dy * dy + dz * dz + dw * dw;
    }
    float bs = block_reduce_sum_256(lsum);
    if (threadIdx.x == 0)
        atomicAdd(qerr, bs * (1.0f / ((float)NT * (float)DH)));
}

__global__ void qsae_zero(float* qerr, float* rerr) {
    if (threadIdx.x == 0) { *qerr = 0.0f; *rerr = 0.0f; }
}

// ---------------------------------------------------------------------------
// Host launcher. Output layout (flat, reference return order):
//   x_rec [NT*DM] | scores [NT*DH] | activations [NT*DH] | qerr [1] | rerr [1]
// ---------------------------------------------------------------------------
extern "C" void kernel_launch(void* const* d_in, const int* in_sizes, int n_in,
                              void* d_out, int out_size, void* d_ws, size_t ws_size,
                              hipStream_t stream) {
    const float* x     = (const float*)d_in[0];
    const float* W_enc = (const float*)d_in[1];
    const float* b_enc = (const float*)d_in[2];
    const float* W_dec = (const float*)d_in[3];

    float* out    = (float*)d_out;
    float* xrec   = out;
    float* scores = xrec + (size_t)NT * DM;
    float* act    = scores + (size_t)NT * DH;
    float* qerr   = act + (size_t)NT * DH;
    float* rerr   = qerr + 1;

    float* thr = (float*)d_ws;   // NT floats = 16 KB scratch

    qsae_zero<<<1, 64, 0, stream>>>(qerr, rerr);
    qsae_encode<<<dim3(NT / 64, DH / 256), 256, 0, stream>>>(x, W_enc, b_enc, scores);
    qsae_thresh<<<NT, 256, 0, stream>>>(scores, thr);
    qsae_act<<<dim3(NT, DH / 4096), 256, 0, stream>>>(scores, thr, act, qerr);
    qsae_decode<<<dim3(NT / 64, DM / 256), 256, 0, stream>>>(act, W_dec, x, xrec, rerr);
}
